// Net_1065151889929
// MI455X (gfx1250) — matrix operand order repaired and checked by reading, not compiled, not run on hardware
//
#include <hip/hip_runtime.h>
#include <hip/hip_bf16.h>

// ---------------------------------------------------------------------------
// Types for CDNA5 WMMA (wave32): v_wmma_f32_16x16x32_bf16
// ---------------------------------------------------------------------------
typedef __attribute__((ext_vector_type(16))) __bf16 bf16x16;
typedef __attribute__((ext_vector_type(8)))  __bf16 bf16x8;
typedef __attribute__((ext_vector_type(8)))  float  f32x8;

// Problem constants
#define BB    2048
#define PP    6
#define OO    10
#define PT    86
#define ROWS_OBJ  (BB * PP * 2 * OO)   // 245760
#define ROWS_IO   (BB * PP)            // 12288
#define NPRIM 40

// ---------------------------------------------------------------------------
// Pack a float weight matrix [Ksrc x Nsrc] (row-major) into a bf16 buffer in
// B-fragment-native layout: tiles of 32(K) x 16(N).  Within a tile, lane l
// (0..31) holds 16 contiguous bf16 covering K = kt*32 + (l>>4)*16 + e,
// N = nt*16 + (l&15)  -> same contiguous K order as the GEMM's A fragments, so the
// GEMM kernel loads each B fragment as 32 contiguous bytes per lane.
// Row remap for reshape-with-padding: padded row k -> group g = k/Gp,
// in-group f = k%Gp; source row = g*Gs + f, valid iff f < Gs.
// (Plain matrices: Gp = Kp, Gs = Ksrc.)
// ---------------------------------------------------------------------------
__global__ void pack_weight_bf16(const float* __restrict__ src,
                                 __bf16* __restrict__ dst,
                                 int Kp, int Np, int Ksrc, int Nsrc,
                                 int Gp, int Gs)
{
    size_t total  = (size_t)Kp * (size_t)Np;
    size_t stride = (size_t)gridDim.x * blockDim.x;
    for (size_t idx = (size_t)blockIdx.x * blockDim.x + threadIdx.x;
         idx < total; idx += stride) {
        int tile = (int)(idx >> 9);       // 512 elems per 32x16 tile
        int w    = (int)(idx & 511);
        int l    = w >> 4;
        int e    = w & 15;
        int ntN  = Np >> 4;
        int kt   = tile / ntN;
        int nt   = tile - kt * ntN;
        int k    = kt * 32 + (l >> 4) * 16 + e;
        int n    = nt * 16 + (l & 15);
        int g    = k / Gp;
        int f    = k - g * Gp;
        float v  = 0.0f;
        if (f < Gs && n < Nsrc) {
            int ks = g * Gs + f;
            if (ks < Ksrc) v = src[(size_t)ks * Nsrc + n];
        }
        dst[idx] = (__bf16)v;
    }
}

// Zero-pad a bias vector (fp32 -> fp32)
__global__ void pad_bias_f32(const float* __restrict__ src,
                             float* __restrict__ dst, int nsrc, int npad)
{
    for (int i = blockIdx.x * blockDim.x + threadIdx.x; i < npad;
         i += gridDim.x * blockDim.x)
        dst[i] = (i < nsrc) ? src[i] : 0.0f;
}

// ---------------------------------------------------------------------------
// Gather + embed.  pix[c,i,j] = wc0*(w_color[c]+b_color) + wc1*(w_x[i]+b_x)
//                             + wc2*(w_y[j]+b_y) + b_comb, times mask.
// Writes padded bf16 activation X0[ROWS_OBJ x 96] (86 valid cols).
// ---------------------------------------------------------------------------
__global__ void gather_embed_bf16(const int* __restrict__ c_idx,
                                  const int* __restrict__ i_idx,
                                  const int* __restrict__ j_idx,
                                  const float* __restrict__ mask,
                                  const float* __restrict__ w_color,
                                  const float* __restrict__ b_color,
                                  const float* __restrict__ w_x,
                                  const float* __restrict__ b_x,
                                  const float* __restrict__ w_y,
                                  const float* __restrict__ b_y,
                                  const float* __restrict__ w_comb,
                                  const float* __restrict__ b_comb,
                                  __bf16* __restrict__ X0)
{
    const float wc0 = w_comb[0], wc1 = w_comb[1], wc2 = w_comb[2];
    const float bco = b_color[0], bxo = b_x[0], byo = b_y[0], bcm = b_comb[0];
    size_t total  = (size_t)ROWS_OBJ * 96;
    size_t stride = (size_t)gridDim.x * blockDim.x;
    for (size_t idx = (size_t)blockIdx.x * blockDim.x + threadIdx.x;
         idx < total; idx += stride) {
        size_t r = idx / 96;
        int    k = (int)(idx - r * 96);
        float  v = 0.0f;
        if (k < PT) {
            size_t s = r * PT + k;
            int c = c_idx[s], i = i_idx[s], j = j_idx[s];
            v = (wc0 * (w_color[c] + bco) +
                 wc1 * (w_x[i] + bxo) +
                 wc2 * (w_y[j] + byo) + bcm) * mask[s];
        }
        X0[idx] = (__bf16)v;
    }
}

// ---------------------------------------------------------------------------
// Generic bf16 WMMA GEMM:  C[M x Np] = A[M x Kp] @ Wp + bias
//   - block = 128 threads = 4 wave32s; block tile = 256(M) x 32(N)
//   - wave tile = 64(M) x 32(N): 4x2 f32x8 accumulators, 8 WMMAs / K-step
//     (4 A-frags + 2 B-frags = 12 x 16B loads per 8 WMMAs -> 1.5 loads/MMA)
//   - A fragments: contiguous K order k = 16*(lane>>4) + e, same as B (two 16B chunks per lane)
//   - B fragments: one 32B contiguous chunk per lane from packed weights
//   - Cb != null : store bf16 padded rows (stride Np), bias fused
//   - Cf != null : final fp32 store, only cols < Nf (stride Nf)
// ---------------------------------------------------------------------------
__global__ __launch_bounds__(128)
void gemm_bf16_wmma(const __bf16* __restrict__ A,
                    const __bf16* __restrict__ Wp,
                    const float*  __restrict__ bias,
                    __bf16* __restrict__ Cb,
                    float*  __restrict__ Cf,
                    int M, int Kp, int Np, int Nf)
{
    const int lane = threadIdx.x & 31;
    const int wave = threadIdx.x >> 5;
    const int l16  = lane & 15;
    const int lhi  = lane >> 4;          // 0 or 1
    const int m0   = blockIdx.x * 256 + wave * 64;
    const int n0   = blockIdx.y * 32;
    const int ntN  = Np >> 4;            // N tiles in packed weights
    const int nt0  = n0 >> 4;

    f32x8 acc[4][2] = {};

    for (int k0 = 0; k0 < Kp; k0 += 32) {
        const int kt = k0 >> 5;
        // Prefetch next K-tile of packed weights into cache (global_prefetch_b8)
        if (k0 + 32 < Kp)
            __builtin_prefetch(Wp + ((size_t)((kt + 1) * ntN + nt0) << 9), 0, 1);

        bf16x16 afrag[4];
        #pragma unroll
        for (int mi = 0; mi < 4; ++mi) {
            const __bf16* ap =
                A + (size_t)(m0 + mi * 16 + l16) * Kp + k0 + lhi * 16;
            bf16x8 lo = *(const bf16x8*)ap;        // K: k0+{0..7} / k0+{16..23}
            bf16x8 hi = *(const bf16x8*)(ap + 8);  // K: k0+{8..15}/k0+{24..31}
            #pragma unroll
            for (int e = 0; e < 8; ++e) { afrag[mi][e] = lo[e]; afrag[mi][8 + e] = hi[e]; }
        }

        bf16x16 bfrag[2];
        #pragma unroll
        for (int ni = 0; ni < 2; ++ni) {
            const __bf16* bp =
                Wp + ((size_t)(kt * ntN + nt0 + ni) << 9) + lane * 16;
            bf16x8 lo = *(const bf16x8*)bp;
            bf16x8 hi = *(const bf16x8*)(bp + 8);
            #pragma unroll
            for (int e = 0; e < 8; ++e) { bfrag[ni][e] = lo[e]; bfrag[ni][8 + e] = hi[e]; }
        }

        #pragma unroll
        for (int mi = 0; mi < 4; ++mi)
            #pragma unroll
            for (int ni = 0; ni < 2; ++ni)
                acc[mi][ni] = __builtin_amdgcn_wmma_f32_16x16x32_bf16(
                    false, afrag[mi], false, bfrag[ni],
                    (short)0, acc[mi][ni], false, false);
    }

    // Epilogue: C/D layout — VGPR r: lanes 0-15 -> M=r, lanes 16-31 -> M=r+8
    #pragma unroll
    for (int ni = 0; ni < 2; ++ni) {
        const int   col = n0 + ni * 16 + l16;
        const float bv  = bias[col];
        #pragma unroll
        for (int mi = 0; mi < 4; ++mi) {
            const int rbase = m0 + mi * 16 + lhi * 8;
            #pragma unroll
            for (int r = 0; r < 8; ++r) {
                float v = acc[mi][ni][r] + bv;
                if (Cf) {
                    if (col < Nf)
                        Cf[(size_t)(rbase + r) * Nf + col] = v;
                } else {
                    Cb[(size_t)(rbase + r) * Np + col] = (__bf16)v;
                }
            }
        }
    }
}

// ---------------------------------------------------------------------------
// Host launcher
// ---------------------------------------------------------------------------
static inline void launch_gemm(const __bf16* A, const __bf16* Wp,
                               const float* bias, __bf16* Cb, float* Cf,
                               int M, int Kp, int Np, int Nf, hipStream_t s)
{
    dim3 grid(M / 256, Np / 32), block(128);
    gemm_bf16_wmma<<<grid, block, 0, s>>>(A, Wp, bias, Cb, Cf, M, Kp, Np, Nf);
}

extern "C" void kernel_launch(void* const* d_in, const int* in_sizes, int n_in,
                              void* d_out, int out_size, void* d_ws, size_t ws_size,
                              hipStream_t stream)
{
    (void)in_sizes; (void)n_in; (void)out_size; (void)ws_size;

    // ---- inputs (setup_inputs order) ----
    const int*   c_idx   = (const int*)  d_in[0];
    const int*   i_idx   = (const int*)  d_in[1];
    const int*   j_idx   = (const int*)  d_in[2];
    const float* mask    = (const float*)d_in[3];
    const float* w_color = (const float*)d_in[4];
    const float* b_color = (const float*)d_in[5];
    const float* w_x     = (const float*)d_in[6];
    const float* b_x     = (const float*)d_in[7];
    const float* w_y     = (const float*)d_in[8];
    const float* b_y     = (const float*)d_in[9];
    const float* w_comb  = (const float*)d_in[10];
    const float* b_comb  = (const float*)d_in[11];
    const float* oW1  = (const float*)d_in[12]; const float* ob1 = (const float*)d_in[13];
    const float* oW2  = (const float*)d_in[14]; const float* ob2 = (const float*)d_in[15];
    const float* oW3  = (const float*)d_in[16]; const float* ob3 = (const float*)d_in[17];
    const float* ioW1 = (const float*)d_in[18]; const float* iob1 = (const float*)d_in[19];
    const float* ioW2 = (const float*)d_in[20]; const float* iob2 = (const float*)d_in[21];
    const float* ioW3 = (const float*)d_in[22]; const float* iob3 = (const float*)d_in[23];
    const float* nW1  = (const float*)d_in[24]; const float* nb1 = (const float*)d_in[25];
    const float* nW2  = (const float*)d_in[26]; const float* nb2 = (const float*)d_in[27];
    const float* nW3  = (const float*)d_in[28]; const float* nb3 = (const float*)d_in[29];
    const float* nW4  = (const float*)d_in[30]; const float* nb4 = (const float*)d_in[31];

    // ---- workspace carving (256B aligned regions) ----
    char* ws = (char*)d_ws;
    size_t off = 0;
    auto take = [&](size_t bytes) -> char* {
        char* p = ws + off;
        off += (bytes + 255) & ~(size_t)255;
        return p;
    };
    // packed bf16 weights
    __bf16* OW1p = (__bf16*)take((size_t)96  * 96   * 2);
    __bf16* OW2p = (__bf16*)take((size_t)96  * 64   * 2);
    __bf16* OW3p = (__bf16*)take((size_t)64  * 64   * 2);
    __bf16* IW1p = (__bf16*)take((size_t)1280 * 1024 * 2);
    __bf16* IW2p = (__bf16*)take((size_t)1024 * 320  * 2);
    __bf16* IW3p = (__bf16*)take((size_t)320  * 320  * 2);
    __bf16* NW1p = (__bf16*)take((size_t)1920 * 1024 * 2);
    __bf16* NW2p = (__bf16*)take((size_t)1024 * 512  * 2);
    __bf16* NW3p = (__bf16*)take((size_t)512  * 256  * 2);
    __bf16* NW4p = (__bf16*)take((size_t)256  * 64   * 2);
    // padded fp32 biases
    float* ob1p  = (float*)take(96   * 4);
    float* ob2p  = (float*)take(64   * 4);
    float* ob3p  = (float*)take(64   * 4);
    float* iob1p = (float*)take(1024 * 4);
    float* iob2p = (float*)take(320  * 4);
    float* iob3p = (float*)take(320  * 4);
    float* nb1p  = (float*)take(1024 * 4);
    float* nb2p  = (float*)take(512  * 4);
    float* nb3p  = (float*)take(256  * 4);
    float* nb4p  = (float*)take(64   * 4);
    // ping-pong activation buffers (each sized for the largest tensor)
    const size_t EBUF = (size_t)ROWS_OBJ * 96; // 23.6M bf16 elems
    __bf16* bufA = (__bf16*)take(EBUF * 2);
    __bf16* bufB = (__bf16*)take(EBUF * 2);

    // ---- weight prep (pack + pad) ----
    dim3 pb(256);
    auto pgrid = [](size_t total) {
        size_t g = (total + 255) / 256;
        return dim3((unsigned)(g > 2048 ? 2048 : g));
    };
    pack_weight_bf16<<<pgrid(96*96),     pb, 0, stream>>>(oW1,  OW1p, 96,   96,  86,   86,   96,   86);
    pack_weight_bf16<<<pgrid(96*64),     pb, 0, stream>>>(oW2,  OW2p, 96,   64,  86,   50,   96,   86);
    pack_weight_bf16<<<pgrid(64*64),     pb, 0, stream>>>(oW3,  OW3p, 64,   64,  50,   50,   64,   50);
    pack_weight_bf16<<<pgrid(1280*1024), pb, 0, stream>>>(ioW1, IW1p, 1280, 1024, 1000, 1000, 64,   50);
    pack_weight_bf16<<<pgrid(1024*320),  pb, 0, stream>>>(ioW2, IW2p, 1024, 320,  1000, 300,  1024, 1000);
    pack_weight_bf16<<<pgrid(320*320),   pb, 0, stream>>>(ioW3, IW3p, 320,  320,  300,  300,  320,  300);
    pack_weight_bf16<<<pgrid(1920*1024), pb, 0, stream>>>(nW1,  NW1p, 1920, 1024, 1800, 1024, 320,  300);
    pack_weight_bf16<<<pgrid(1024*512),  pb, 0, stream>>>(nW2,  NW2p, 1024, 512,  1024, 512,  1024, 1024);
    pack_weight_bf16<<<pgrid(512*256),   pb, 0, stream>>>(nW3,  NW3p, 512,  256,  512,  256,  512,  512);
    pack_weight_bf16<<<pgrid(256*64),    pb, 0, stream>>>(nW4,  NW4p, 256,  64,   256,  40,   256,  256);

    pad_bias_f32<<<1, 256, 0, stream>>>(ob1,  ob1p,  86,   96);
    pad_bias_f32<<<1, 256, 0, stream>>>(ob2,  ob2p,  50,   64);
    pad_bias_f32<<<1, 256, 0, stream>>>(ob3,  ob3p,  50,   64);
    pad_bias_f32<<<4, 256, 0, stream>>>(iob1, iob1p, 1000, 1024);
    pad_bias_f32<<<2, 256, 0, stream>>>(iob2, iob2p, 300,  320);
    pad_bias_f32<<<2, 256, 0, stream>>>(iob3, iob3p, 300,  320);
    pad_bias_f32<<<4, 256, 0, stream>>>(nb1,  nb1p,  1024, 1024);
    pad_bias_f32<<<2, 256, 0, stream>>>(nb2,  nb2p,  512,  512);
    pad_bias_f32<<<1, 256, 0, stream>>>(nb3,  nb3p,  256,  256);
    pad_bias_f32<<<1, 256, 0, stream>>>(nb4,  nb4p,  40,   64);

    // ---- gather/embed: X0 = bufA [245760 x 96] bf16 ----
    gather_embed_bf16<<<4096, 256, 0, stream>>>(
        c_idx, i_idx, j_idx, mask, w_color, b_color, w_x, b_x, w_y, b_y,
        w_comb, b_comb, bufA);

    // ---- GEMM chain (ping-pong bufA <-> bufB) ----
    // obj_fe
    launch_gemm(bufA, OW1p, ob1p,  bufB, nullptr, ROWS_OBJ, 96,   96,  0, stream);
    launch_gemm(bufB, OW2p, ob2p,  bufA, nullptr, ROWS_OBJ, 96,   64,  0, stream);
    launch_gemm(bufA, OW3p, ob3p,  bufB, nullptr, ROWS_OBJ, 64,   64,  0, stream);
    // io_fe: bufB viewed as [12288 x 1280] (20 objs x 64 padded feats)
    launch_gemm(bufB, IW1p, iob1p, bufA, nullptr, ROWS_IO,  1280, 1024, 0, stream);
    launch_gemm(bufA, IW2p, iob2p, bufB, nullptr, ROWS_IO,  1024, 320,  0, stream);
    launch_gemm(bufB, IW3p, iob3p, bufA, nullptr, ROWS_IO,  320,  320,  0, stream);
    // head: bufA viewed as [2048 x 1920] (6 pairs x 320 padded feats)
    launch_gemm(bufA, NW1p, nb1p,  bufB, nullptr, BB,       1920, 1024, 0, stream);
    launch_gemm(bufB, NW2p, nb2p,  bufA, nullptr, BB,       1024, 512,  0, stream);
    launch_gemm(bufA, NW3p, nb3p,  bufB, nullptr, BB,       512,  256,  0, stream);
    // final: fp32 store to d_out [2048 x 40]
    launch_gemm(bufB, NW4p, nb4p,  nullptr, (float*)d_out, BB, 256, 64, NPRIM, stream);
}
